// TimeWiseCVAE_45440753992297
// MI455X (gfx1250) — matrix operand order repaired and checked by reading, not compiled, not run on hardware
//
#include <hip/hip_runtime.h>
#include <math.h>

// ---------------------------------------------------------------------------
// TimeWiseCVAE forward for MI455X (gfx1250, wave32, WMMA + TDM), v3.
// conv1d layers run as im2col GEMMs on v_wmma_f32_16x16x32_f16. Activations
// are f16 channel-fastest rows (padded to multiples of 32). Weights are
// pre-packed into B-tile-major f16 panels; each wave DMAs its panel column
// into LDS with tensor_load_to_lds (TDM) and reads B via ds_load, keeping the
// vector-memory pipe free for the streaming A loads.
// ---------------------------------------------------------------------------

typedef __attribute__((ext_vector_type(16))) _Float16 v16h;
typedef __attribute__((ext_vector_type(8)))  _Float16 v8h;
typedef __attribute__((ext_vector_type(8)))  float    v8f;
typedef __attribute__((ext_vector_type(4)))  unsigned int tdm_u32x4;
typedef __attribute__((ext_vector_type(8)))  int          tdm_i32x8;
typedef __attribute__((ext_vector_type(4)))  int          tdm_i32x4;

#define SRATE  48000
#define TLEN   62400
#define LSTEPS 975
#define NH     32
#define BATCH  8

#if __has_builtin(__builtin_amdgcn_tensor_load_to_lds) && \
    __has_builtin(__builtin_amdgcn_s_wait_tensorcnt)
#define HAS_TDM 1
#else
#define HAS_TDM 0
#endif

#if HAS_TDM
// Issue one TDM 2D tile load: nchunks rows of 512 f16 at stride row_stride_el
// halves, into this workgroup's LDS at offset 0.
__device__ __forceinline__ void tdm_load_panel(const _Float16* gsrc,
                                               int nchunks, int row_stride_el)
{
  tdm_u32x4 g0 = {};
  tdm_i32x8 g1 = {};
  tdm_i32x4 g2 = {}, g3 = {};
  unsigned long long ga = (unsigned long long)(uintptr_t)gsrc;
  g0[0] = 1u;                                           // count=1, user D#
  g0[1] = 0u;                                           // lds_addr = 0 (sP at 0)
  g0[2] = (unsigned)(ga & 0xffffffffu);                 // global_addr[31:0]
  g0[3] = (unsigned)((ga >> 32) & 0x01ffffffu) | (2u << 30); // addr[56:32]|type=2
  g1[0] = 1 << 16;                                      // data_size = 2 bytes
  g1[1] = (int)((512u & 0xffffu) << 16);                // tensor_dim0[15:0]
  g1[2] = (int)((512u >> 16) | ((unsigned)(nchunks & 0xffff) << 16)); // dim0hi|dim1lo
  g1[3] = (int)((((unsigned)nchunks >> 16) & 0xffffu) | (512u << 16)); // dim1hi|tile_dim0
  g1[4] = nchunks & 0xffff;                             // tile_dim1 (tile_dim2=0)
  g1[5] = row_stride_el;                                // tensor_dim0_stride[31:0]
  g1[6] = 0; g1[7] = 0;
#if defined(__clang_major__) && (__clang_major__ >= 23)
  tdm_i32x8 g4 = {};
  __builtin_amdgcn_tensor_load_to_lds(g0, g1, g2, g3, g4, 0);
#else
  __builtin_amdgcn_tensor_load_to_lds(g0, g1, g2, g3, 0);
#endif
}
#endif

// ---------------------------------------------------------------------------
// WMMA conv. Reduction index space r = tap*CIP + ci (CIP % 32 == 0).
// A tile (16x32 f16): lane<16 -> M=lane,    K = {0..7, 16..23} of chunk
//                     lane>=16 -> M=lane-16, K = {8..15, 24..31} of chunk
// B panel pre-packed so each lane reads 16 contiguous halves (from LDS).
// ACT: 0=none 1=leaky_relu(0.2) 2=relu 3=sigmoid 5=clip(-10,2)
// ---------------------------------------------------------------------------
template<int ACT, int F16OUT, int CIP, int K>
__global__ __launch_bounds__(32)
void conv_wmma2_k(const _Float16* __restrict__ in, int in_len, int row_stride,
                  const _Float16* __restrict__ panel,
                  const float* __restrict__ bias,
                  void* __restrict__ outv, int out_len, int CO,
                  int stride, int pad_left)
{
  constexpr int NCHUNK = K * (CIP / 32);
  __shared__ _Float16 sP[NCHUNK * 512];    // only LDS object -> offset 0

  const int  lane = threadIdx.x;
  const int  l15  = lane & 15;
  const bool hi   = lane >= 16;
  const int  m0   = blockIdx.x << 4;
  const int  nt   = blockIdx.y;
  const int  b    = blockIdx.z;
  const int  t_out  = m0 + l15;
  const int  ntiles = CO >> 4;
  const _Float16* inb = in + (size_t)b * in_len * row_stride;

#if HAS_TDM
  tdm_load_panel(panel + (size_t)nt * 512, NCHUNK, ntiles * 512);
#endif
  __builtin_prefetch(in, 0, 1);

#if HAS_TDM
  __builtin_amdgcn_s_wait_tensorcnt(0);
#else
  // fallback: copy panel column to LDS with plain loads
  for (int c = 0; c < NCHUNK; ++c)
    *(v16h*)(sP + c * 512 + lane * 16) =
      *(const v16h*)(panel + ((size_t)c * ntiles + nt) * 512 + lane * 16);
  __builtin_amdgcn_s_waitcnt(0);
#endif

  v8f acc = {};
#pragma unroll
  for (int tap = 0; tap < K; ++tap) {
    const int t_in = t_out * stride + tap - pad_left;
    const bool ok  = (t_in >= 0) & (t_in < in_len) & (t_out < out_len);
    const _Float16* rowp = inb + (size_t)(ok ? t_in : 0) * row_stride;
    const int chunkBase = tap * (CIP >> 5);
#pragma unroll 2
    for (int cb = 0; cb < CIP; cb += 32) {
      v8h a_lo = {}, a_hi = {};
      if (ok) {
        const int base = cb + (hi ? 8 : 0);
        a_lo = *(const v8h*)(rowp + base);        // K {0..7} / {8..15}
        a_hi = *(const v8h*)(rowp + base + 16);   // K {16..23} / {24..31}
      }
      v16h A = __builtin_shufflevector(a_lo, a_hi,
                 0,1,2,3,4,5,6,7,8,9,10,11,12,13,14,15);
      const int chunk = chunkBase + (cb >> 5);
      v16h B = *(const v16h*)(sP + chunk * 512 + lane * 16);   // ds_load_b128 x2
      acc = __builtin_amdgcn_wmma_f32_16x16x32_f16(false, A, false, B,
                                                   (short)0, acc, false, false);
    }
  }

  const int n = (nt << 4) + l15;
  const float bv = bias[n];
#pragma unroll
  for (int j = 0; j < 8; ++j) {
    const int t = m0 + (hi ? 8 + j : j);   // C/D: lanes0-15 M=j, lanes16-31 M=8+j
    if (t < out_len) {
      float v = acc[j] + bv;
      if      (ACT == 1) v = v > 0.f ? v : 0.2f * v;
      else if (ACT == 2) v = fmaxf(v, 0.f);
      else if (ACT == 3) v = 1.f / (1.f + __expf(-v));
      else if (ACT == 5) v = fminf(fmaxf(v, -10.f), 2.f);
      const size_t o = ((size_t)b * out_len + t) * CO + n;
      if (F16OUT) ((_Float16*)outv)[o] = (_Float16)v;
      else        ((float*)outv)[o]    = v;
    }
  }
}

// One-shot weight re-layout: (K,CI,CO) f32 -> B-tile-major f16 panel.
// panel[chunk][ntile][lane*16+j] holds K = chunk*32 + (lane<16 ? 0 : 8) + (j<8 ? j : 8+j),
// N = nt*16 + (lane&15); zeros where ci >= CI (CIP padding).
__global__ __launch_bounds__(32)
void pack_w_k(const float* __restrict__ w, _Float16* __restrict__ panel,
              int K, int CI, int CIP, int CO)
{
  const int lane = threadIdx.x;
  const int chunk = blockIdx.x, nt = blockIdx.y;
  const int n = (nt << 4) + (lane & 15);
  const int ntiles = CO >> 4;
  _Float16* dst = panel + ((size_t)chunk * ntiles + nt) * 512 + lane * 16;
#pragma unroll
  for (int j = 0; j < 16; ++j) {
    const int kk = ((lane < 16) ? 0 : 8) + (j < 8 ? j : 8 + j);
    const int r = chunk * 32 + kk;
    const int tap = r / CIP, ci = r - tap * CIP;
    const float v = (ci < CI) ? w[((size_t)tap * CI + ci) * CO + n] : 0.f;
    dst[j] = (_Float16)v;
  }
}

// Pack a 1-channel f32 signal (+ optional 4 cond channels) into f16 [B,T,32].
__global__ void pack_in_k(const float* __restrict__ src, const float* __restrict__ cond,
                          _Float16* __restrict__ dst)
{
  int idx = blockIdx.x * blockDim.x + threadIdx.x;
  if (idx >= BATCH * TLEN) return;
  int b = idx / TLEN;
  _Float16* row = dst + (size_t)idx * 32;
  row[0] = (_Float16)src[idx];
  if (cond) {
    row[1] = (_Float16)cond[b*4+0]; row[2] = (_Float16)cond[b*4+1];
    row[3] = (_Float16)cond[b*4+2]; row[4] = (_Float16)cond[b*4+3];
  } else { row[1] = (_Float16)0.f; row[2] = (_Float16)0.f;
           row[3] = (_Float16)0.f; row[4] = (_Float16)0.f; }
#pragma unroll
  for (int i = 5; i < 32; ++i) row[i] = (_Float16)0.f;
}

// z = mean + exp(0.5*logvar)*eps, written as padded f16 rows [B,975,96]:
// ch 0..63 = z, 64..67 = cond, 68..95 = 0  (z_cond concat baked in).
__global__ void reparam_pad_k(const float* __restrict__ mean, const float* __restrict__ logvar,
                              const float* __restrict__ eps, const float* __restrict__ cond,
                              _Float16* __restrict__ zp)
{
  int idx = blockIdx.x * blockDim.x + threadIdx.x;
  if (idx >= BATCH * LSTEPS * 96) return;
  int c = idx % 96, bt = idx / 96, b = bt / LSTEPS;
  float v;
  if (c < 64) { int s = bt * 64 + c; v = mean[s] + __expf(0.5f * logvar[s]) * eps[s]; }
  else if (c < 68) v = cond[b * 4 + (c - 64)];
  else v = 0.f;
  zp[idx] = (_Float16)v;
}

// CO == 1 conv layers (env/nenv/nfilt finals), f16 input, stride 1.
template<int ACT>
__global__ void conv1d_co1_k(const _Float16* __restrict__ in, int len, int CI,
                             const float* __restrict__ w, const float* __restrict__ bias,
                             float* __restrict__ out, int K, int pad_left)
{
  int idx = blockIdx.x * blockDim.x + threadIdx.x;
  if (idx >= BATCH * len) return;
  int b = idx / len, t = idx - b * len;
  const _Float16* inb = in + (size_t)b * len * CI;
  float acc = bias[0];
  for (int tap = 0; tap < K; ++tap) {
    int ti = t + tap - pad_left;
    if (ti < 0 || ti >= len) continue;
    const _Float16* row = inb + (size_t)ti * CI;
    for (int ci = 0; ci < CI; ++ci) acc += (float)row[ci] * w[tap * CI + ci];
  }
  if      (ACT == 3) acc = 1.f / (1.f + __expf(-acc));
  else if (ACT == 4) acc = tanhf(acc);
  out[idx] = acc;
}

__global__ void timbre_k(const float* __restrict__ cond,
                         const float* __restrict__ w_s, const float* __restrict__ b_s,
                         const float* __restrict__ w_a, const float* __restrict__ b_a,
                         const float* __restrict__ w_p, const float* __restrict__ b_p,
                         float* __restrict__ combined, float* __restrict__ scal)
{
  int tid = threadIdx.x;
  int b = tid >> 5, h = tid & 31;
  if (b >= BATCH) return;
  float pitch = cond[b*4+0], sw = cond[b*4+1], aw = cond[b*4+2], pw = cond[b*4+3];
  float ds = 1.f / (1.f + __expf(-(sw * w_s[h] + b_s[h])));
  float da = 1.f / (1.f + __expf(-(aw * w_a[h] + b_a[h])));
  float dp = 1.f / (1.f + __expf(-(pw * w_p[h] + b_p[h])));
  combined[b * NH + h] = ds * sw + da * aw + dp * pw;
  if (h == 0) {
    scal[b] = 440.f * exp2f((pitch * 35.f + 36.f - 69.f) / 12.f);
    float nr = sw * 0.5f + aw * 0.2f + pw * 0.05f;
    scal[BATCH + b]     = nr;
    scal[2 * BATCH + b] = 1.f - nr * 0.5f;
  }
}

// Fused linear resize (half-pixel) + harmonic sin-sum + envelope/noise + tanh.
__global__ void synth_k(const _Float16* __restrict__ amps,  // [B,975,32] f16
                        const float* __restrict__ env,      // [B,975]
                        const float* __restrict__ nenv,     // [B,975]
                        const float* __restrict__ nf,       // [B,T]
                        const float* __restrict__ combined,
                        const float* __restrict__ scal,
                        float* __restrict__ out)
{
  int idx = blockIdx.x * blockDim.x + threadIdx.x;
  if (idx >= BATCH * TLEN) return;
  int b = idx / TLEN, t = idx - b * TLEN;
  float src = ((float)t + 0.5f) * ((float)LSTEPS / (float)TLEN) - 0.5f;
  float sf = floorf(src);
  int   i0 = (int)sf;
  float wg = src - sf;
  int ia = min(max(i0, 0), LSTEPS - 1);
  int ic = min(max(i0 + 1, 0), LSTEPS - 1);
  float e  = env [b*LSTEPS+ia]*(1.f-wg) + env [b*LSTEPS+ic]*wg;
  float ne = nenv[b*LSTEPS+ia]*(1.f-wg) + nenv[b*LSTEPS+ic]*wg;
  float f0 = scal[b], nr = scal[BATCH + b], hr = scal[2*BATCH + b];
  const _Float16* a0 = amps + ((size_t)b * LSTEPS + ia) * NH;
  const _Float16* a1 = amps + ((size_t)b * LSTEPS + ic) * NH;
  float w0 = 6.28318530718f * f0 * ((float)t / (float)SRATE);
  float wave = 0.f;
#pragma unroll 8
  for (int h = 0; h < NH; ++h) {
    float amp = (float)a0[h] * (1.f - wg) + (float)a1[h] * wg;
    float shaped = amp * (0.5f + 1.5f * combined[b * NH + h]);
    wave += shaped * __sinf(w0 * (float)(h + 1));
  }
  out[idx] = tanhf(wave * e * hr + ne * nf[idx] * nr);
}

extern "C" void kernel_launch(void* const* d_in, const int* in_sizes, int n_in,
                              void* d_out, int out_size, void* d_ws, size_t ws_size,
                              hipStream_t stream) {
  // Inputs: x, cond, eps, random_noise, then params in JAX pytree order
  // (dict keys sorted): enc[4], enc_logvar, enc_mean, env[3], ha[4],
  // nenv[3], nfilt[3], ts{acid,pluck,screech}; each (w, b).
  const float* x    = (const float*)d_in[0];
  const float* cond = (const float*)d_in[1];
  const float* eps  = (const float*)d_in[2];
  const float* rnd  = (const float*)d_in[3];
  const float *enc0_w=(const float*)d_in[4],  *enc0_b=(const float*)d_in[5];
  const float *enc1_w=(const float*)d_in[6],  *enc1_b=(const float*)d_in[7];
  const float *enc2_w=(const float*)d_in[8],  *enc2_b=(const float*)d_in[9];
  const float *enc3_w=(const float*)d_in[10], *enc3_b=(const float*)d_in[11];
  const float *lv_w  =(const float*)d_in[12], *lv_b  =(const float*)d_in[13];
  const float *mn_w  =(const float*)d_in[14], *mn_b  =(const float*)d_in[15];
  const float *env0_w=(const float*)d_in[16], *env0_b=(const float*)d_in[17];
  const float *env1_w=(const float*)d_in[18], *env1_b=(const float*)d_in[19];
  const float *env2_w=(const float*)d_in[20], *env2_b=(const float*)d_in[21];
  const float *ha0_w =(const float*)d_in[22], *ha0_b =(const float*)d_in[23];
  const float *ha1_w =(const float*)d_in[24], *ha1_b =(const float*)d_in[25];
  const float *ha2_w =(const float*)d_in[26], *ha2_b =(const float*)d_in[27];
  const float *ha3_w =(const float*)d_in[28], *ha3_b =(const float*)d_in[29];
  const float *ne0_w =(const float*)d_in[30], *ne0_b =(const float*)d_in[31];
  const float *ne1_w =(const float*)d_in[32], *ne1_b =(const float*)d_in[33];
  const float *ne2_w =(const float*)d_in[34], *ne2_b =(const float*)d_in[35];
  const float *nf0_w =(const float*)d_in[36], *nf0_b =(const float*)d_in[37];
  const float *nf1_w =(const float*)d_in[38], *nf1_b =(const float*)d_in[39];
  const float *nf2_w =(const float*)d_in[40], *nf2_b =(const float*)d_in[41];
  const float *ts_a_w=(const float*)d_in[42], *ts_a_b=(const float*)d_in[43];
  const float *ts_p_w=(const float*)d_in[44], *ts_p_b=(const float*)d_in[45];
  const float *ts_s_w=(const float*)d_in[46], *ts_s_b=(const float*)d_in[47];

  // ---- workspace: f16 activations (offsets in halves, all 16B aligned) ----
  _Float16* hb = (_Float16*)d_ws;
  _Float16* H0  = hb;               // [B,T,32] packed input (x+cond, later rnd)
  _Float16* E1  = hb + 16000000;    // 8*15600*64
  _Float16* E2  = hb + 24000000;    // 8*3900*128
  _Float16* E3  = hb + 28000000;    // 8*1950*256
  _Float16* E4  = hb + 32000000;    // 8*975*512
  _Float16* NF1 = hb + 36000000;    // 8*62400*32
  _Float16* NF2 = hb + 52000000;    // 8*62400*16
  _Float16* ZP  = hb + 60000000;    // 8*975*96 (z + cond + zero pad)
  _Float16* HA1 = hb + 60800000;    // 8*975*128
  _Float16* HA2 = hb + 61800000;
  _Float16* HA3 = hb + 62800000;    // 8*975*64
  _Float16* AMPS= hb + 63300000;    // 8*975*32
  _Float16* EV1 = hb + 63600000;    // 8*975*64
  _Float16* EV2 = hb + 64100000;    // 8*975*32
  _Float16* NE1 = hb + 64400000;    // 8*975*64
  _Float16* NE2 = hb + 64900000;    // 8*975*32
  // weight panels (f16, B-tile-major)
  _Float16* P_E0 = hb + 65200000;   // enc1: 5 chunks * 4 nt
  _Float16* P_E1 = hb + 65220000;   // enc2: 10 * 8
  _Float16* P_E2 = hb + 65270000;   // enc3: 20 * 16
  _Float16* P_E3 = hb + 65440000;   // enc4: 24 * 32
  _Float16* P_MN = hb + 65840000;   // mean: 48 * 4
  _Float16* P_LV = hb + 65940000;   // logvar: 48 * 4
  _Float16* P_H0 = hb + 66040000;   // ha1: 15 * 8
  _Float16* P_H1 = hb + 66110000;   // ha2: 20 * 8
  _Float16* P_H2 = hb + 66200000;   // ha3: 20 * 4
  _Float16* P_H3 = hb + 66250000;   // ha4: 10 * 2
  _Float16* P_V0 = hb + 66270000;   // env1: 10 * 4
  _Float16* P_V1 = hb + 66300000;   // env2: 10 * 2
  _Float16* P_N0 = hb + 66320000;   // nenv1: 15 * 4
  _Float16* P_N1 = hb + 66350000;   // nenv2: 10 * 2
  _Float16* P_F0 = hb + 66370000;   // nfilt1: 5 * 2
  _Float16* P_F1 = hb + 66380000;   // nfilt2: 5 * 1
  // f32 region
  float* fb  = (float*)(hb + 66400000);
  float* ENV = fb;                  // 8*975
  float* NENV= fb + 10000;          // 8*975
  float* NFO = fb + 20000;          // 8*62400
  float* COMB= fb + 520000;         // 8*32
  float* SCAL= fb + 521000;         // 24

  float* OUT = (float*)d_out;                 // [B,T,1]
  float* ZM  = OUT + BATCH * TLEN;            // z_mean   [B,975,64] f32
  float* ZLV = ZM  + BATCH * LSTEPS * 64;     // z_logvar [B,975,64] f32

  dim3 blk(32);
  // ---- pack all weight panels (tiny, once per call) ----
  pack_w_k<<<dim3( 5, 4), blk, 0, stream>>>(enc0_w, P_E0, 5, 5,   32,  64);
  pack_w_k<<<dim3(10, 8), blk, 0, stream>>>(enc1_w, P_E1, 5, 64,  64,  128);
  pack_w_k<<<dim3(20,16), blk, 0, stream>>>(enc2_w, P_E2, 5, 128, 128, 256);
  pack_w_k<<<dim3(24,32), blk, 0, stream>>>(enc3_w, P_E3, 3, 256, 256, 512);
  pack_w_k<<<dim3(48, 4), blk, 0, stream>>>(mn_w,   P_MN, 3, 512, 512, 64);
  pack_w_k<<<dim3(48, 4), blk, 0, stream>>>(lv_w,   P_LV, 3, 512, 512, 64);
  pack_w_k<<<dim3(15, 8), blk, 0, stream>>>(ha0_w,  P_H0, 5, 68,  96,  128);
  pack_w_k<<<dim3(20, 8), blk, 0, stream>>>(ha1_w,  P_H1, 5, 128, 128, 128);
  pack_w_k<<<dim3(20, 4), blk, 0, stream>>>(ha2_w,  P_H2, 5, 128, 128, 64);
  pack_w_k<<<dim3(10, 2), blk, 0, stream>>>(ha3_w,  P_H3, 5, 64,  64,  32);
  pack_w_k<<<dim3(10, 4), blk, 0, stream>>>(env0_w, P_V0, 5, 64,  64,  64);
  pack_w_k<<<dim3(10, 2), blk, 0, stream>>>(env1_w, P_V1, 5, 64,  64,  32);
  pack_w_k<<<dim3(15, 4), blk, 0, stream>>>(ne0_w,  P_N0, 5, 68,  96,  64);
  pack_w_k<<<dim3(10, 2), blk, 0, stream>>>(ne1_w,  P_N1, 5, 64,  64,  32);
  pack_w_k<<<dim3( 5, 2), blk, 0, stream>>>(nf0_w,  P_F0, 5, 1,   32,  32);
  pack_w_k<<<dim3( 5, 1), blk, 0, stream>>>(nf1_w,  P_F1, 5, 1,   32,  16);

  const int nBT = (BATCH * TLEN + 255) / 256;
  // ---- encoder (leaky_relu 0.2), SAME pad lefts: 0,0,1,0 ----
  pack_in_k<<<nBT, 256, 0, stream>>>(x, cond, H0);
  conv_wmma2_k<1,1,32,5> <<<dim3(975, 4, BATCH), blk, 0, stream>>>(H0, TLEN, 32,  P_E0, enc0_b, E1, 15600, 64,  4, 0);
  conv_wmma2_k<1,1,64,5> <<<dim3(244, 8, BATCH), blk, 0, stream>>>(E1, 15600, 64, P_E1, enc1_b, E2, 3900, 128,  4, 0);
  conv_wmma2_k<1,1,128,5><<<dim3(122,16, BATCH), blk, 0, stream>>>(E2, 3900, 128, P_E2, enc2_b, E3, 1950, 256,  2, 1);
  conv_wmma2_k<1,1,256,3><<<dim3(61, 32, BATCH), blk, 0, stream>>>(E3, 1950, 256, P_E3, enc3_b, E4, 975,  512,  2, 0);
  // ---- z_mean (f32, none), z_logvar (f32, clip), K=3 s=1 pad 1 ----
  conv_wmma2_k<0,0,512,3><<<dim3(61, 4, BATCH), blk, 0, stream>>>(E4, 975, 512, P_MN, mn_b, ZM,  975, 64, 1, 1);
  conv_wmma2_k<5,0,512,3><<<dim3(61, 4, BATCH), blk, 0, stream>>>(E4, 975, 512, P_LV, lv_b, ZLV, 975, 64, 1, 1);
  {
    int n = BATCH * LSTEPS * 96;
    reparam_pad_k<<<(n + 255) / 256, 256, 0, stream>>>(ZM, ZLV, eps, cond, ZP);
  }
  // ---- harmonic-amp decoder, K=5 pad 2 ----
  conv_wmma2_k<2,1,96,5> <<<dim3(61, 8, BATCH), blk, 0, stream>>>(ZP,  LSTEPS, 96,  P_H0, ha0_b, HA1, LSTEPS, 128, 1, 2);
  conv_wmma2_k<2,1,128,5><<<dim3(61, 8, BATCH), blk, 0, stream>>>(HA1, LSTEPS, 128, P_H1, ha1_b, HA2, LSTEPS, 128, 1, 2);
  conv_wmma2_k<2,1,128,5><<<dim3(61, 4, BATCH), blk, 0, stream>>>(HA2, LSTEPS, 128, P_H2, ha2_b, HA3, LSTEPS, 64,  1, 2);
  conv_wmma2_k<3,1,64,5> <<<dim3(61, 2, BATCH), blk, 0, stream>>>(HA3, LSTEPS, 64,  P_H3, ha3_b, AMPS, LSTEPS, 32, 1, 2);
  // ---- envelope head (reads only z channels of ZP via CIP=64 < row_stride=96) ----
  conv_wmma2_k<2,1,64,5> <<<dim3(61, 4, BATCH), blk, 0, stream>>>(ZP,  LSTEPS, 96, P_V0, env0_b, EV1, LSTEPS, 64, 1, 2);
  conv_wmma2_k<2,1,64,5> <<<dim3(61, 2, BATCH), blk, 0, stream>>>(EV1, LSTEPS, 64, P_V1, env1_b, EV2, LSTEPS, 32, 1, 2);
  conv1d_co1_k<3><<<(BATCH * LSTEPS + 255) / 256, 256, 0, stream>>>(EV2, LSTEPS, 32, env2_w, env2_b, ENV, 5, 2);
  // ---- noise-envelope head (full 96-ch z_cond) ----
  conv_wmma2_k<2,1,96,5> <<<dim3(61, 4, BATCH), blk, 0, stream>>>(ZP,  LSTEPS, 96, P_N0, ne0_b, NE1, LSTEPS, 64, 1, 2);
  conv_wmma2_k<2,1,64,5> <<<dim3(61, 2, BATCH), blk, 0, stream>>>(NE1, LSTEPS, 64, P_N1, ne1_b, NE2, LSTEPS, 32, 1, 2);
  conv1d_co1_k<3><<<(BATCH * LSTEPS + 255) / 256, 256, 0, stream>>>(NE2, LSTEPS, 32, ne2_w, ne2_b, NENV, 5, 2);
  // ---- noise filter on random_noise (H0 region reused after enc1 consumed it) ----
  pack_in_k<<<nBT, 256, 0, stream>>>(rnd, nullptr, H0);
  conv_wmma2_k<2,1,32,5> <<<dim3(3900, 2, BATCH), blk, 0, stream>>>(H0,  TLEN, 32, P_F0, nf0_b, NF1, TLEN, 32, 1, 2);
  conv_wmma2_k<2,1,32,5> <<<dim3(3900, 1, BATCH), blk, 0, stream>>>(NF1, TLEN, 32, P_F1, nf1_b, NF2, TLEN, 16, 1, 2);
  conv1d_co1_k<4><<<nBT, 256, 0, stream>>>(NF2, TLEN, 16, nf2_w, nf2_b, NFO, 5, 2);
  // ---- timbre gates + fused resize/harmonic/tanh synthesis ----
  timbre_k<<<1, 256, 0, stream>>>(cond, ts_s_w, ts_s_b, ts_a_w, ts_a_b, ts_p_w, ts_p_b, COMB, SCAL);
  synth_k<<<nBT, 256, 0, stream>>>(AMPS, ENV, NENV, NFO, COMB, SCAL, OUT);
  (void)in_sizes; (void)n_in; (void)out_size; (void)ws_size;
}
